// DeepSeekBlock_65807488910002
// MI455X (gfx1250) — compile-verified, matrix operand order checked by reading, not run on hardware
//
#include <hip/hip_runtime.h>
#include <hip/hip_bf16.h>
#include <math.h>

// ---------------------------------------------------------------------------
// Problem constants (from reference)
// ---------------------------------------------------------------------------
constexpr int Bc   = 2;
constexpr int Sc   = 2048;
constexpr int Dc   = 1024;
constexpr int Hc   = 16;
constexpr int KVHc = 4;
constexpr int HDc  = 64;      // head dim
constexpr int QL   = 64;      // q latent
constexpr int KVL  = 32;      // kv latent
constexpr int Ec   = 8;       // experts
constexpr int FFc  = 4096;    // ffn dim
constexpr int BS   = Bc * Sc;             // 4096 tokens
constexpr int CAP  = 2 * BS + Ec * 128;   // slot capacity, segments pad to 128 -> 9216
constexpr int MAXMB = CAP / 128;          // 72 max 128-row blocks per expert

// Macro-tile geometry for LDS-tiled GEMMs: block = 256 thr = 8 waves,
// tile 128(M) x 64(N) x 32(K); wave grid 4(M) x 2(N), 32x32 per wave.
constexpr int BM = 128, BN = 64, BK = 32;

typedef _Float16 half_t;
typedef __attribute__((ext_vector_type(16))) _Float16 v16h;
typedef __attribute__((ext_vector_type(8)))  float    v8f;
typedef int v4i_ __attribute__((vector_size(16)));    // async-LDS builtin operand type

// ---------------------------------------------------------------------------
// gfx1250 async global->LDS path (ASYNCcnt), with sync fallback.
// Builtin signature (probe-confirmed): (v4i as1*, v4i as3*, Ii, Ii).
// ---------------------------------------------------------------------------
#if defined(__has_builtin)
#if __has_builtin(__builtin_amdgcn_global_load_async_to_lds_b128) && \
    __has_builtin(__builtin_amdgcn_s_wait_asynccnt)
#define ATH_ASYNC 1
#endif
#endif
#ifndef ATH_ASYNC
#define ATH_ASYNC 0
#endif

__device__ __forceinline__ void cp16B_g2lds(const void* g, void* l) {
#if ATH_ASYNC
  __builtin_amdgcn_global_load_async_to_lds_b128(
      (__attribute__((address_space(1))) v4i_*)(g),
      (__attribute__((address_space(3))) v4i_*)(l), 0, 0);
#else
  *(uint4*)l = *(const uint4*)g;
#endif
}
__device__ __forceinline__ void async_wait_all() {
#if ATH_ASYNC
  __builtin_amdgcn_s_wait_asynccnt(0);
#endif
}

// ---------------------------------------------------------------------------
// WMMA helpers.  CDNA5 16x16x32 f16 -> f32 accumulate (ISA 7.12.2, wave32):
//  A (16x32): lane L row (L&15); elem i -> K = 8*(L>>4) + (i<8 ? i : i+8)
//  B (32x16): lane L col (L&15); elem i -> K = 8*(L>>4) + (i<8 ? i : i+8)
//  C (16x16): lane L col (L&15); VGPR ii -> row ii + 8*(L>>4)
// ---------------------------------------------------------------------------
__device__ __forceinline__ v8f wmma_f16(v16h a, v16h b, v8f c) {
  return __builtin_amdgcn_wmma_f32_16x16x32_f16(false, a, false, b, (short)0, c,
                                                false, false);
}
// p -> &X[row*ld + lk0] (two contiguous 8-elem runs at +0 and +16)
__device__ __forceinline__ v16h frag_f32(const float* p) {
  v16h r;
#pragma unroll
  for (int i = 0; i < 8; ++i) { r[i] = (_Float16)p[i]; r[i + 8] = (_Float16)p[16 + i]; }
  return r;
}
__device__ __forceinline__ v16h frag_f16(const half_t* p) {
  v16h r;
#pragma unroll
  for (int i = 0; i < 8; ++i) { r[i] = p[i]; r[i + 8] = p[16 + i]; }
  return r;
}
// strided B-frag straight from global f32 (small-GEMM path only)
__device__ __forceinline__ v16h bfrag_g_f32(const float* p, int ldb) {
  v16h r;
#pragma unroll
  for (int i = 0; i < 8; ++i) {
    r[i]     = (_Float16)p[(size_t)i * ldb];
    r[i + 8] = (_Float16)p[(size_t)(16 + i) * ldb];
  }
  return r;
}

// ---------------------------------------------------------------------------
// Workspace init
// ---------------------------------------------------------------------------
__global__ void init_kernel(int* counts, int* fill, int* row_token, float* moe) {
  const size_t gid = (size_t)blockIdx.x * blockDim.x + threadIdx.x;
  if (gid < Ec) { counts[gid] = 0; fill[gid] = 0; }
  if (gid < CAP) row_token[gid] = -1;
  if (gid < (size_t)BS * Dc) moe[gid] = 0.0f;
}

// ---------------------------------------------------------------------------
// LayerNorm over D=1024.  MODE 0: Out = LN(X).  MODE 1: Out = Res + LN(X).
// ---------------------------------------------------------------------------
template <int MODE>
__global__ void ln_kernel(const float* __restrict__ X, const float* __restrict__ G,
                          const float* __restrict__ Bb, const float* __restrict__ Res,
                          float* __restrict__ Out) {
  const int t   = blockIdx.x;
  const int tid = threadIdx.x;
  const float* x = X + (size_t)t * Dc;
  __shared__ float red[256];

  float s = 0.0f;
  for (int i = tid; i < Dc; i += 256) s += x[i];
  red[tid] = s; __syncthreads();
  for (int off = 128; off > 0; off >>= 1) {
    if (tid < off) red[tid] += red[tid + off];
    __syncthreads();
  }
  const float mean = red[0] * (1.0f / Dc);
  __syncthreads();

  float v = 0.0f;
  for (int i = tid; i < Dc; i += 256) { float d = x[i] - mean; v += d * d; }
  red[tid] = v; __syncthreads();
  for (int off = 128; off > 0; off >>= 1) {
    if (tid < off) red[tid] += red[tid + off];
    __syncthreads();
  }
  const float rstd = rsqrtf(red[0] * (1.0f / Dc) + 1e-5f);

  for (int i = tid; i < Dc; i += 256) {
    float y = (x[i] - mean) * rstd * G[i] + Bb[i];
    if (MODE == 1) y += Res[(size_t)t * Dc + i];
    Out[(size_t)t * Dc + i] = y;
  }
}

// ---------------------------------------------------------------------------
// Simple wave-per-tile GEMM, only for tiny-N projections (kv_a, N=32).
// ---------------------------------------------------------------------------
__global__ void gemm16_small(const float* __restrict__ A, const float* __restrict__ Bw,
                             float* __restrict__ C, int M, int N, int K) {
  const int wave = (int)((blockIdx.x * (size_t)blockDim.x + threadIdx.x) >> 5);
  const int lane = threadIdx.x & 31;
  const int ntiles = N >> 4;
  const int mt = wave / ntiles;
  const int nt = wave - mt * ntiles;
  if (mt * 16 >= M) return;
  const int lk0 = (lane >> 4) * 8;
  const int col = lane & 15;
  const int n   = nt * 16 + col;
  const size_t arow = (size_t)(mt * 16 + col) * K;
  v8f acc = {};
  for (int k0 = 0; k0 < K; k0 += 32) {
    v16h af = frag_f32(A + arow + k0 + lk0);
    v16h bf = bfrag_g_f32(Bw + (size_t)(k0 + lk0) * N + n, N);
    acc = wmma_f16(af, bf, acc);
  }
#pragma unroll
  for (int ii = 0; ii < 8; ++ii)
    C[(size_t)(mt * 16 + ii + lk0) * N + n] = acc[ii];
}

// ---------------------------------------------------------------------------
// LDS-tiled dense GEMM: C[MxN] = A[MxK] @ B[KxN] (+bias) (+Res).
// A panel staged raw via async global->LDS; B panel converted to f16 and
// transposed (n-major) so fragments load as contiguous ds_b128 runs.
// EPI 0: plain (+bias if given).  EPI 1: + bias + Res.
// ---------------------------------------------------------------------------
template <int EPI, bool AHALF>
__global__ void gemm_tiled(const void* __restrict__ A_, const float* __restrict__ Bw,
                           const float* __restrict__ bias, const float* __restrict__ Res,
                           float* __restrict__ C, int M, int N, int K) {
  __shared__ alignas(16) char smem[BM * BK * 4 + BN * BK * 2];
  float*  Asf = (float*)smem;
  half_t* Ash = (half_t*)smem;
  half_t* Bs  = (half_t*)(smem + BM * BK * 4);

  const int nbm = M / BM;
  const int bm  = blockIdx.x % nbm;
  const int bn  = blockIdx.x / nbm;

  const int tid    = threadIdx.x;
  const int lane   = tid & 31;
  const int wid    = tid >> 5;
  const int wave_m = wid & 3;        // 0..3 -> 32-row strip
  const int wave_n = wid >> 2;       // 0..1 -> 32-col strip
  const int lk0    = (lane >> 4) * 8;
  const int col    = lane & 15;

  const int arow_l = tid >> 1;             // 0..127
  const int akp    = (tid & 1) * 16;
  const size_t arow_g = (size_t)(bm * BM + arow_l);

  const int bk_l = tid >> 3;               // 0..31
  const int bn_l = (tid & 7) * 8;          // 0..56
  const int n_g0 = bn * BN;

  v8f acc[4]; { const v8f z = {}; acc[0]=z; acc[1]=z; acc[2]=z; acc[3]=z; }

  for (int k0 = 0; k0 < K; k0 += BK) {
    if (AHALF) {
      const half_t* src = (const half_t*)A_ + arow_g * K + k0 + akp;
      half_t* dst = Ash + arow_l * BK + akp;
      cp16B_g2lds(src, dst);
      cp16B_g2lds(src + 8, dst + 8);
    } else {
      const float* src = (const float*)A_ + arow_g * K + k0 + akp;
      float* dst = Asf + arow_l * BK + akp;
      cp16B_g2lds(src, dst);
      cp16B_g2lds(src + 4, dst + 4);
      cp16B_g2lds(src + 8, dst + 8);
      cp16B_g2lds(src + 12, dst + 12);
    }
    {
      const float* src = Bw + (size_t)(k0 + bk_l) * N + n_g0 + bn_l;
#pragma unroll
      for (int j = 0; j < 8; ++j)
        Bs[(bn_l + j) * BK + bk_l] = (_Float16)src[j];
    }
    async_wait_all();
    __syncthreads();

    v16h af0, af1;
    if (AHALF) {
      af0 = frag_f16(Ash + (wave_m * 32 + col) * BK + lk0);
      af1 = frag_f16(Ash + (wave_m * 32 + 16 + col) * BK + lk0);
    } else {
      af0 = frag_f32(Asf + (wave_m * 32 + col) * BK + lk0);
      af1 = frag_f32(Asf + (wave_m * 32 + 16 + col) * BK + lk0);
    }
    const v16h bf0 = frag_f16(Bs + (wave_n * 32 + col) * BK + lk0);
    const v16h bf1 = frag_f16(Bs + (wave_n * 32 + 16 + col) * BK + lk0);
    acc[0] = wmma_f16(af0, bf0, acc[0]);
    acc[1] = wmma_f16(af0, bf1, acc[1]);
    acc[2] = wmma_f16(af1, bf0, acc[2]);
    acc[3] = wmma_f16(af1, bf1, acc[3]);
    __syncthreads();
  }

  const int row_base = bm * BM + wave_m * 32;
  const int n_base   = n_g0 + wave_n * 32;
#pragma unroll
  for (int mt2 = 0; mt2 < 2; ++mt2)
#pragma unroll
    for (int nt2 = 0; nt2 < 2; ++nt2) {
      const v8f a = acc[mt2 * 2 + nt2];
      const int n = n_base + nt2 * 16 + col;
      const float bv = bias ? bias[n] : 0.0f;
#pragma unroll
      for (int ii = 0; ii < 8; ++ii) {
        const size_t m = (size_t)(row_base + mt2 * 16 + ii + lk0);
        float cv = a[ii] + bv;
        if (EPI == 1) cv += Res[m * N + n];
        C[m * N + n] = cv;
      }
    }
}

// ---------------------------------------------------------------------------
// RoPE on q and k; split kv into K (roped, f16) and V-transposed (f16).
// ---------------------------------------------------------------------------
__global__ void rope_kernel(const float* __restrict__ Qf, const float* __restrict__ KVf,
                            half_t* __restrict__ Q16, half_t* __restrict__ K16,
                            half_t* __restrict__ Vt16) {
  const size_t gid = (size_t)blockIdx.x * blockDim.x + threadIdx.x;
  const int HT = Hc + KVHc;                         // 20
  if (gid >= (size_t)BS * HT * 32) return;
  const int i  = (int)(gid & 31);
  const int hh = (int)((gid >> 5) % HT);
  const size_t ts = gid / ((size_t)32 * HT);
  const int s = (int)(ts % Sc);
  const int b = (int)(ts / Sc);

  const float freq = (float)s * expf(-logf(10000.0f) * (float)(2 * i) / (float)HDc);
  const float cs = cosf(freq), sn = sinf(freq);

  if (hh < Hc) {
    const float* q = Qf + (ts * Hc + hh) * HDc;
    half_t* o = Q16 + (ts * Hc + hh) * HDc;
    const float x1 = q[2 * i], x2 = q[2 * i + 1];
    o[2 * i]     = (_Float16)(x1 * cs - x2 * sn);
    o[2 * i + 1] = (_Float16)(x1 * sn + x2 * cs);
  } else {
    const int kvh = hh - Hc;
    const float* kv = KVf + (ts * KVHc + kvh) * (HDc * 2);
    const float k1 = kv[(2 * i) * 2 + 0], k2 = kv[(2 * i + 1) * 2 + 0];
    const float v1 = kv[(2 * i) * 2 + 1], v2 = kv[(2 * i + 1) * 2 + 1];
    half_t* ko = K16 + (ts * KVHc + kvh) * HDc;
    ko[2 * i]     = (_Float16)(k1 * cs - k2 * sn);
    ko[2 * i + 1] = (_Float16)(k1 * sn + k2 * cs);
    const size_t vb = ((size_t)b * KVHc + kvh) * HDc * Sc;   // Vt: (b,kvh,hd,S)
    Vt16[vb + (size_t)(2 * i) * Sc + s]     = (_Float16)v1;
    Vt16[vb + (size_t)(2 * i + 1) * Sc + s] = (_Float16)v2;
  }
}

// ---------------------------------------------------------------------------
// Causal flash attention, block-cooperative: one block = 8 consecutive 16-row
// q tiles of one (b,h).  Each 32-key chunk of K (32x64) and V^T (64x32) is
// staged once per block in LDS (async global->LDS), then all 8 waves consume
// it.  St(32k x 16q) = K @ Q^T so softmax reduces in C-frag M dim; exp(St)
// becomes the PV B-frag in place.  O^T(64hd x 16q) += V^T @ P^T.
// Waves past their causal limit skip compute (wave-uniform) but keep barriers.
// ---------------------------------------------------------------------------
__global__ void attn_kernel(const half_t* __restrict__ Q, const half_t* __restrict__ Kb,
                            const half_t* __restrict__ Vt, half_t* __restrict__ O) {
  __shared__ alignas(16) half_t Ks[32 * HDc];   // [key][hd]   4KB
  __shared__ alignas(16) half_t Vs[HDc * 32];   // [hd][key]   4KB

  const int QBLK = Sc / 128;                    // 16 q-blocks per (b,h)
  const int qb0 = blockIdx.x % QBLK;
  const int h   = (blockIdx.x / QBLK) % Hc;
  const int b   = blockIdx.x / (QBLK * Hc);
  const int kvh = h / (Hc / KVHc);

  const int tid  = threadIdx.x;
  const int lane = tid & 31;
  const int wid  = tid >> 5;
  const int qt   = qb0 * 8 + wid;               // this wave's q tile
  const int lk0  = (lane >> 4) * 8;
  const int col  = lane & 15;
  const int qpos = qt * 16 + col;

  // K-panel loader: 32 rows x 64 halfs = 256 16B chunks, one per thread
  const int krow_l = tid >> 3, koff = (tid & 7) * 8;
  // V-panel loader: 64 rows x 32 halfs = 256 16B chunks, one per thread
  const int vrow_l = tid >> 2, voff = (tid & 3) * 8;
  const half_t* VtBase = Vt + ((size_t)b * KVHc + kvh) * HDc * Sc;

  // Q^T B-fragments (per wave, from global; two 32-wide hd halves)
  v16h qb[2];
  {
    const half_t* qr = Q + (((size_t)b * Sc + qpos) * Hc + h) * HDc;
#pragma unroll
    for (int hf = 0; hf < 2; ++hf) {
      const half_t* p = qr + hf * 32 + lk0;
#pragma unroll
      for (int i = 0; i < 8; ++i) { qb[hf][i] = p[i]; qb[hf][i + 8] = p[16 + i]; }
    }
  }

  const v8f vz = {};
  v8f of[4]; of[0] = vz; of[1] = vz; of[2] = vz; of[3] = vz;
  float mrun = -3.0e38f, lrun = 0.0f;

  const int myLast = (qt * 16 + 15) >> 5;           // this wave's causal chunk limit
  const int cmax   = ((qb0 * 8 + 7) * 16 + 15) >> 5; // block max (last wave's limit)

  for (int c = 0; c <= cmax; ++c) {
    const int kbase = c * 32;

    // ---- stage K chunk (32x64) and V^T chunk (64x32) in LDS ----
    cp16B_g2lds(Kb + (((size_t)b * Sc + kbase + krow_l) * KVHc + kvh) * HDc + koff,
                Ks + krow_l * HDc + koff);
    cp16B_g2lds(VtBase + (size_t)vrow_l * Sc + kbase + voff,
                Vs + vrow_l * 32 + voff);
    async_wait_all();
    __syncthreads();

    if (c <= myLast) {                               // wave-uniform
      // scores^T: two 16-key tiles, hd reduced in two K=32 WMMAs each
      v8f st[2]; st[0] = vz; st[1] = vz;
#pragma unroll
      for (int t = 0; t < 2; ++t) {
        const half_t* kr = Ks + (t * 16 + col) * HDc;
#pragma unroll
        for (int hf = 0; hf < 2; ++hf) {
          const v16h ka = frag_f16(kr + hf * 32 + lk0);
          st[t] = wmma_f16(ka, qb[hf], st[t]);
        }
      }

      // scale + causal mask + online softmax (column q = lane&15)
      float sv[16];
      float smax = -3.0e38f;
#pragma unroll
      for (int t = 0; t < 2; ++t)
#pragma unroll
        for (int ii = 0; ii < 8; ++ii) {
          const int key = kbase + t * 16 + lk0 + ii;
          float s = st[t][ii] * 0.125f;              // HD^-0.5
          if (key > qpos) s = -3.0e38f;
          sv[t * 8 + ii] = s;
          smax = fmaxf(smax, s);
        }
      smax = fmaxf(smax, __shfl_xor(smax, 16, 32));
      const float mnew  = fmaxf(mrun, smax);
      const float alpha = __expf(mrun - mnew);
      float rsum = 0.0f;
      v16h pb;                                       // P^T B-frag, maps 1:1 to sv[]
#pragma unroll
      for (int j = 0; j < 16; ++j) {
        const float p = __expf(sv[j] - mnew);
        rsum += p;
        pb[j] = (_Float16)p;
      }
      rsum += __shfl_xor(rsum, 16, 32);
      lrun = lrun * alpha + rsum;
      mrun = mnew;
#pragma unroll
      for (int t4 = 0; t4 < 4; ++t4) of[t4] *= alpha;

      // O^T += V^T @ P^T (4 hd tiles, K = 32 keys)
#pragma unroll
      for (int t4 = 0; t4 < 4; ++t4) {
        const v16h va = frag_f16(Vs + (t4 * 16 + col) * 32 + lk0);
        of[t4] = wmma_f16(va, pb, of[t4]);
      }
    }
    __syncthreads();
  }

  const float inv = 1.0f / lrun;
  half_t* orow = O + (((size_t)b * Sc + qpos) * Hc + h) * HDc;
#pragma unroll
  for (int t4 = 0; t4 < 4; ++t4)
#pragma unroll
    for (int ii = 0; ii < 8; ++ii)
      orow[t4 * 16 + lk0 + ii] = (_Float16)(of[t4][ii] * inv);
}

// ---------------------------------------------------------------------------
// Router logits (N=8 -> plain VALU dot products), straight into d_out.
// ---------------------------------------------------------------------------
__global__ void router_kernel(const float* __restrict__ H2, const float* __restrict__ Rw,
                              float* __restrict__ Lg) {
  const int gid = blockIdx.x * blockDim.x + threadIdx.x;
  if (gid >= BS * Ec) return;
  const int e = gid & (Ec - 1);
  const int t = gid >> 3;
  const float* hrow = H2 + (size_t)t * Dc;
  float s = 0.0f;
  for (int k = 0; k < Dc; ++k) s = fmaf(hrow[k], Rw[k * Ec + e], s);
  Lg[gid] = s;
}

// ---------------------------------------------------------------------------
// Top-2 gating (first-occurrence ties like lax.top_k) + expert bucketing.
// ---------------------------------------------------------------------------
__global__ void top2_kernel(const float* __restrict__ Lg, int* __restrict__ slot_e,
                            float* __restrict__ slot_g, int* __restrict__ counts) {
  const int t = blockIdx.x * blockDim.x + threadIdx.x;
  if (t >= BS) return;
  float v0 = -3.0e38f, v1 = -3.0e38f;
  int i0 = 0, i1 = 0;
  for (int e = 0; e < Ec; ++e) {
    const float l = Lg[t * Ec + e];
    if (l > v0)      { v1 = v0; i1 = i0; v0 = l; i0 = e; }
    else if (l > v1) { v1 = l; i1 = e; }
  }
  const float e1 = __expf(v1 - v0);
  const float denom = 1.0f / (1.0f + e1);
  slot_e[2 * t]     = i0;  slot_g[2 * t]     = denom;
  slot_e[2 * t + 1] = i1;  slot_g[2 * t + 1] = e1 * denom;
  atomicAdd(&counts[i0], 1);
  atomicAdd(&counts[i1], 1);
}

__global__ void scan_kernel(const int* __restrict__ counts, int* __restrict__ offsets,
                            int* __restrict__ fill) {
  if (blockIdx.x == 0 && threadIdx.x == 0) {
    int acc = 0;
    for (int e = 0; e < Ec; ++e) {
      offsets[e] = acc;
      acc += (counts[e] + 127) & ~127;     // pad segments to 128-row blocks
      fill[e] = 0;
    }
    offsets[Ec] = acc;
  }
}

__global__ void assign_kernel(const int* __restrict__ slot_e, const float* __restrict__ slot_g,
                              const int* __restrict__ offsets, int* __restrict__ fill,
                              int* __restrict__ row_token, float* __restrict__ row_gate) {
  const int sidx = blockIdx.x * blockDim.x + threadIdx.x;
  if (sidx >= 2 * BS) return;
  const int e = slot_e[sidx];
  const int pos = atomicAdd(&fill[e], 1);
  const int row = offsets[e] + pos;
  row_token[row] = sidx >> 1;
  row_gate[row]  = slot_g[sidx];
}

// ---------------------------------------------------------------------------
// Grouped MoE GEMM #1 (LDS-tiled): hidden = gelu(gather(h2) @ fc_w[e] + b).
// ---------------------------------------------------------------------------
__global__ void moe_fc_tiled(const float* __restrict__ H2, const float* __restrict__ FCw,
                             const float* __restrict__ FCb, half_t* __restrict__ Hid,
                             const int* __restrict__ offsets, const int* __restrict__ row_token) {
  __shared__ alignas(16) char smem[BM * BK * 4 + BN * BK * 2];
  float*  Asf = (float*)smem;
  half_t* Bs  = (half_t*)(smem + BM * BK * 4);

  const int nbn = FFc / BN;                           // 64
  const int nb  = blockIdx.x % nbn;
  const int rest = blockIdx.x / nbn;
  const int mb  = rest % MAXMB;
  const int e   = rest / MAXMB;
  if (e >= Ec) return;
  const int base = offsets[e], end = offsets[e + 1];
  const int row0 = base + mb * BM;
  if (row0 >= end) return;                            // block-uniform

  const int tid = threadIdx.x;
  const int lane = tid & 31;
  const int wid = tid >> 5;
  const int wave_m = wid & 3, wave_n = wid >> 2;
  const int lk0 = (lane >> 4) * 8, col = lane & 15;

  const int arow_l = tid >> 1;
  const int akp    = (tid & 1) * 16;
  const int tok    = row_token[row0 + arow_l];        // -1 on padding
  const float* Ag  = H2 + (size_t)(tok < 0 ? 0 : tok) * Dc;
  const float* Bg  = FCw + (size_t)e * Dc * FFc;
  const int bk_l = tid >> 3, bn_l = (tid & 7) * 8;
  const int n_g0 = nb * BN;

  v8f acc[4]; { const v8f z = {}; acc[0]=z; acc[1]=z; acc[2]=z; acc[3]=z; }

  for (int k0 = 0; k0 < Dc; k0 += BK) {
    float* dst = Asf + arow_l * BK + akp;
    if (tok >= 0) {
      const float* src = Ag + k0 + akp;
      cp16B_g2lds(src, dst);
      cp16B_g2lds(src + 4, dst + 4);
      cp16B_g2lds(src + 8, dst + 8);
      cp16B_g2lds(src + 12, dst + 12);
    } else {
#pragma unroll
      for (int j = 0; j < 16; ++j) dst[j] = 0.0f;
    }
    {
      const float* src = Bg + (size_t)(k0 + bk_l) * FFc + n_g0 + bn_l;
#pragma unroll
      for (int j = 0; j < 8; ++j)
        Bs[(bn_l + j) * BK + bk_l] = (_Float16)src[j];
    }
    async_wait_all();
    __syncthreads();

    const v16h af0 = frag_f32(Asf + (wave_m * 32 + col) * BK + lk0);
    const v16h af1 = frag_f32(Asf + (wave_m * 32 + 16 + col) * BK + lk0);
    const v16h bf0 = frag_f16(Bs + (wave_n * 32 + col) * BK + lk0);
    const v16h bf1 = frag_f16(Bs + (wave_n * 32 + 16 + col) * BK + lk0);
    acc[0] = wmma_f16(af0, bf0, acc[0]);
    acc[1] = wmma_f16(af0, bf1, acc[1]);
    acc[2] = wmma_f16(af1, bf0, acc[2]);
    acc[3] = wmma_f16(af1, bf1, acc[3]);
    __syncthreads();
  }

  const int row_base = row0 + wave_m * 32;
  const int n_base   = n_g0 + wave_n * 32;
#pragma unroll
  for (int mt2 = 0; mt2 < 2; ++mt2)
#pragma unroll
    for (int nt2 = 0; nt2 < 2; ++nt2) {
      const v8f a = acc[mt2 * 2 + nt2];
      const int n = n_base + nt2 * 16 + col;
      const float bv = FCb[e * FFc + n];
#pragma unroll
      for (int ii = 0; ii < 8; ++ii) {
        const float xv = a[ii] + bv;
        const float gl = 0.5f * xv * (1.0f + erff(xv * 0.70710678f));  // exact gelu
        Hid[(size_t)(row_base + mt2 * 16 + ii + lk0) * FFc + n] = (_Float16)gl;
      }
    }
}

// ---------------------------------------------------------------------------
// Grouped MoE GEMM #2 (LDS-tiled): y = hidden @ proj_w[e] + b; gated scatter.
// Exactly 2 commutative f32 atomic adds per output element -> deterministic.
// ---------------------------------------------------------------------------
__global__ void moe_proj_tiled(const half_t* __restrict__ Hid, const float* __restrict__ Pw,
                               const float* __restrict__ Pb, float* __restrict__ Moe,
                               const int* __restrict__ offsets, const int* __restrict__ row_token,
                               const float* __restrict__ row_gate) {
  __shared__ alignas(16) char smem[BM * BK * 2 + BN * BK * 2];
  half_t* Ash = (half_t*)smem;
  half_t* Bs  = (half_t*)(smem + BM * BK * 2);

  const int nbn = Dc / BN;                            // 16
  const int nb  = blockIdx.x % nbn;
  const int rest = blockIdx.x / nbn;
  const int mb  = rest % MAXMB;
  const int e   = rest / MAXMB;
  if (e >= Ec) return;
  const int base = offsets[e], end = offsets[e + 1];
  const int row0 = base + mb * BM;
  if (row0 >= end) return;

  const int tid = threadIdx.x;
  const int lane = tid & 31;
  const int wid = tid >> 5;
  const int wave_m = wid & 3, wave_n = wid >> 2;
  const int lk0 = (lane >> 4) * 8, col = lane & 15;

  const int arow_l = tid >> 1;
  const int akp    = (tid & 1) * 16;
  const half_t* Ag = Hid + (size_t)(row0 + arow_l) * FFc;
  const float* Bg  = Pw + (size_t)e * FFc * Dc;
  const int bk_l = tid >> 3, bn_l = (tid & 7) * 8;
  const int n_g0 = nb * BN;

  v8f acc[4]; { const v8f z = {}; acc[0]=z; acc[1]=z; acc[2]=z; acc[3]=z; }

  for (int k0 = 0; k0 < FFc; k0 += BK) {
    {
      const half_t* src = Ag + k0 + akp;
      half_t* dst = Ash + arow_l * BK + akp;
      cp16B_g2lds(src, dst);
      cp16B_g2lds(src + 8, dst + 8);
    }
    {
      const float* src = Bg + (size_t)(k0 + bk_l) * Dc + n_g0 + bn_l;
#pragma unroll
      for (int j = 0; j < 8; ++j)
        Bs[(bn_l + j) * BK + bk_l] = (_Float16)src[j];
    }
    async_wait_all();
    __syncthreads();

    const v16h af0 = frag_f16(Ash + (wave_m * 32 + col) * BK + lk0);
    const v16h af1 = frag_f16(Ash + (wave_m * 32 + 16 + col) * BK + lk0);
    const v16h bf0 = frag_f16(Bs + (wave_n * 32 + col) * BK + lk0);
    const v16h bf1 = frag_f16(Bs + (wave_n * 32 + 16 + col) * BK + lk0);
    acc[0] = wmma_f16(af0, bf0, acc[0]);
    acc[1] = wmma_f16(af0, bf1, acc[1]);
    acc[2] = wmma_f16(af1, bf0, acc[2]);
    acc[3] = wmma_f16(af1, bf1, acc[3]);
    __syncthreads();
  }

  const int row_base = row0 + wave_m * 32;
  const int n_base   = n_g0 + wave_n * 32;
#pragma unroll
  for (int mt2 = 0; mt2 < 2; ++mt2)
#pragma unroll
    for (int nt2 = 0; nt2 < 2; ++nt2) {
      const v8f a = acc[mt2 * 2 + nt2];
      const int n = n_base + nt2 * 16 + col;
      const float bv = Pb[e * Dc + n];
#pragma unroll
      for (int ii = 0; ii < 8; ++ii) {
        const int r = row_base + mt2 * 16 + ii + lk0;
        const int tok = row_token[r];
        if (tok < 0) continue;
        atomicAdd(&Moe[(size_t)tok * Dc + n], (a[ii] + bv) * row_gate[r]);
      }
    }
}

// ---------------------------------------------------------------------------
// Host launch
// ---------------------------------------------------------------------------
extern "C" void kernel_launch(void* const* d_in, const int* in_sizes, int n_in,
                              void* d_out, int out_size, void* d_ws, size_t ws_size,
                              hipStream_t stream) {
  const float* x       = (const float*)d_in[0];
  const float* ln1_g   = (const float*)d_in[1];
  const float* ln1_b   = (const float*)d_in[2];
  const float* q_a_w   = (const float*)d_in[3];
  const float* q_b_w   = (const float*)d_in[4];
  const float* kv_a_w  = (const float*)d_in[5];
  const float* kv_b_w  = (const float*)d_in[6];
  const float* out_w   = (const float*)d_in[7];
  const float* out_b   = (const float*)d_in[8];
  const float* ln2_g   = (const float*)d_in[9];
  const float* ln2_b   = (const float*)d_in[10];
  const float* router_w= (const float*)d_in[11];
  const float* fc_w    = (const float*)d_in[12];
  const float* fc_b    = (const float*)d_in[13];
  const float* proj_w  = (const float*)d_in[14];
  const float* proj_b  = (const float*)d_in[15];
  const float* mln_g   = (const float*)d_in[16];
  const float* mln_b   = (const float*)d_in[17];

  float* out_x   = (float*)d_out;                    // [BS, D]
  float* out_log = (float*)d_out + (size_t)BS * Dc;  // [BS, E]

  // --- workspace carve (256B aligned) ---
  char* w = (char*)d_ws;
  auto carve = [&](size_t bytes) -> void* {
    void* p = (void*)w;
    w += (bytes + 255) & ~(size_t)255;
    return p;
  };
  float*  h      = (float*) carve((size_t)BS * Dc * 4);
  float*  qlat   = (float*) carve((size_t)BS * QL * 4);
  float*  qf     = (float*) carve((size_t)BS * Dc * 4);
  float*  kvlat  = (float*) carve((size_t)BS * KVL * 4);
  float*  kvf    = (float*) carve((size_t)BS * KVHc * HDc * 2 * 4);
  half_t* q16    = (half_t*)carve((size_t)BS * Dc * 2);
  half_t* k16    = (half_t*)carve((size_t)BS * KVHc * HDc * 2);
  half_t* vt16   = (half_t*)carve((size_t)BS * KVHc * HDc * 2);
  half_t* o16    = (half_t*)carve((size_t)BS * Dc * 2);
  float*  x2     = (float*) carve((size_t)BS * Dc * 4);
  float*  h2     = (float*) carve((size_t)BS * Dc * 4);
  float*  moe    = (float*) carve((size_t)BS * Dc * 4);
  int*    slot_e = (int*)   carve((size_t)2 * BS * 4);
  float*  slot_g = (float*) carve((size_t)2 * BS * 4);
  int*    counts = (int*)   carve(Ec * 4);
  int*    fill   = (int*)   carve(Ec * 4);
  int*    offs   = (int*)   carve((Ec + 1) * 4);
  int*    row_tok= (int*)   carve((size_t)CAP * 4);
  float*  row_g  = (float*) carve((size_t)CAP * 4);
  half_t* hidden = (half_t*)carve((size_t)CAP * FFc * 2);
  (void)ws_size; (void)n_in; (void)in_sizes; (void)out_size;

  const int TB = 256;
  auto tiled_blocks = [](int M, int N) { return (M / BM) * (N / BN); };

  // 0) init counters / padding markers / MoE accumulator
  init_kernel<<<((size_t)BS * Dc + TB - 1) / TB, TB, 0, stream>>>(counts, fill, row_tok, moe);

  // 1) LN1
  ln_kernel<0><<<BS, TB, 0, stream>>>(x, ln1_g, ln1_b, nullptr, h);

  // 2) low-rank projections (WMMA; tiled where N >= 64)
  gemm_tiled<0, false><<<tiled_blocks(BS, QL), TB, 0, stream>>>(h, q_a_w, nullptr, nullptr, qlat, BS, QL, Dc);
  gemm_tiled<0, false><<<tiled_blocks(BS, Dc), TB, 0, stream>>>(qlat, q_b_w, nullptr, nullptr, qf, BS, Dc, QL);
  gemm16_small<<<((BS / 16) * (KVL / 16) * 32 + TB - 1) / TB, TB, 0, stream>>>(h, kv_a_w, kvlat, BS, KVL, Dc);
  gemm_tiled<0, false><<<tiled_blocks(BS, KVHc * HDc * 2), TB, 0, stream>>>(kvlat, kv_b_w, nullptr, nullptr,
                                                                            kvf, BS, KVHc * HDc * 2, KVL);

  // 3) RoPE + split K/V (V transposed for the PV WMMA)
  {
    const size_t nthr = (size_t)BS * (Hc + KVHc) * 32;
    rope_kernel<<<(nthr + TB - 1) / TB, TB, 0, stream>>>(qf, kvf, q16, k16, vt16);
  }

  // 4) causal flash attention (block-cooperative, K/V chunks staged in LDS)
  attn_kernel<<<Bc * Hc * (Sc / 128), TB, 0, stream>>>(q16, k16, vt16, o16);

  // 5) out projection + bias + residual (tiled WMMA, f16 A)
  gemm_tiled<1, true><<<tiled_blocks(BS, Dc), TB, 0, stream>>>(o16, out_w, out_b, x, x2, BS, Dc, Dc);

  // 6) LN2
  ln_kernel<0><<<BS, TB, 0, stream>>>(x2, ln2_g, ln2_b, nullptr, h2);

  // 7) router logits straight into d_out
  router_kernel<<<(BS * Ec + TB - 1) / TB, TB, 0, stream>>>(h2, router_w, out_log);

  // 8) top-2 gating + expert bucketing
  top2_kernel<<<(BS + TB - 1) / TB, TB, 0, stream>>>(out_log, slot_e, slot_g, counts);
  scan_kernel<<<1, 1, 0, stream>>>(counts, offs, fill);
  assign_kernel<<<(2 * BS + TB - 1) / TB, TB, 0, stream>>>(slot_e, slot_g, offs, fill, row_tok, row_g);

  // 9) grouped MoE FFN (tiled WMMA): gelu(h2@fc) -> hidden f16; hidden@proj gated scatter
  moe_fc_tiled<<<Ec * MAXMB * (FFc / BN), TB, 0, stream>>>(h2, fc_w, fc_b, hidden, offs, row_tok);
  moe_proj_tiled<<<Ec * MAXMB * (Dc / BN), TB, 0, stream>>>(hidden, proj_w, proj_b, moe, offs, row_tok, row_g);

  // 10) final: out = x2 + LN(moe)
  ln_kernel<1><<<BS, TB, 0, stream>>>(moe, mln_g, mln_b, x2, out_x);
}
